// Alignment_Block_23321672417448
// MI455X (gfx1250) — matrix operand order repaired and checked by reading, not compiled, not run on hardware
//
#include <hip/hip_runtime.h>
#include <hip/hip_bf16.h>
#include <math.h>

// ---------------------------------------------------------------------------
// Problem constants (from reference)
// ---------------------------------------------------------------------------
#define BB 32
#define TT 181
#define DD 512
#define HH 8
#define HDD 64
#define DEPTH 6
#define FFN 2048
#define ROWS (BB * TT)          // 5792
#define ROWS_P 5888             // padded to multiple of 128
#define TPAD 192                // padded scores row stride (multiple of 32, cols zeroed)
#define SROWS 256               // padded scores rows per (b,h) slice (multiple of 128)
#define LN_EPS 1e-5f

typedef __attribute__((ext_vector_type(16))) _Float16 v16h;
typedef __attribute__((ext_vector_type(2)))  __fp16   v2fp16;
typedef __attribute__((ext_vector_type(8)))  float    v8f;

// pack two f32 -> packed f16 pair (v_cvt_pk_rtz_f16_f32)
static __device__ __forceinline__ unsigned pk2(float a, float b) {
    v2fp16 h = __builtin_amdgcn_cvt_pkrtz(a, b);
    return __builtin_bit_cast(unsigned, h);
}

// ---------------------------------------------------------------------------
// Batched WMMA GEMM:  C = act(alpha * (A @ B) + bias + biasMat)
//   Guard-free staging (padded workspace buffers, K mult of 32, ld mult of 4).
//   Software-pipelined: global loads of tile k+1 issue before the WMMAs of
//   tile k; LDS double buffer; ONE barrier per K-step.
//   block: 256 threads (8 wave32), tile 128x64, K-step 32,
//   wave tile 32x32 = 2x2 v_wmma_f32_16x16x32_f16.
// ---------------------------------------------------------------------------
#define GTM 128
#define GTN 64
#define GTK 32
#define LDA_T 48   // halves per aT row (96B: 32B-aligned rows)
#define LDB_T 48   // halves per bT row

// A-column swizzle: K halves stored as [0..7,16..23 | 8..15,24..31] so the
// fragment for lane-group g is the contiguous half-range [16g, 16g+16).
static __device__ __forceinline__ int aswz(int k) {
    return (k & 7) | ((k >> 4) << 3) | (((k >> 3) & 1) << 4);
}

__global__ __launch_bounds__(256)
void gemm_wmma_kernel(const float* __restrict__ A, const float* __restrict__ Bm,
                      const float* __restrict__ bias, const float* __restrict__ biasMat,
                      float* __restrict__ C,
                      int M, int N, int K,
                      int lda, int ldb, int ldc, int ldbm,
                      long long sAb, long long sAh,
                      long long sBb, long long sBh,
                      long long sCb, long long sCh,
                      int nbatch2, int transB, int act, float alpha)
{
    __shared__ _Float16 aT[2][GTM][LDA_T];   // double-buffered, swizzled columns
    __shared__ _Float16 bT[2][GTN][LDB_T];   // double-buffered, transposed [n][k]

    const int z  = blockIdx.z;
    const int zb = z / nbatch2;
    const int zh = z - zb * nbatch2;
    A  += zb * sAb + zh * sAh;
    Bm += zb * sBb + zh * sBh;
    C  += zb * sCb + zh * sCh;

    const int mBlock = blockIdx.y * GTM;
    const int nBlock = blockIdx.x * GTN;
    const int tid  = threadIdx.x;
    const int lane = tid & 31;
    const int wave = tid >> 5;
    const int wM = (wave & 3) * 32;   // 4 waves along M
    const int wN = (wave >> 2) * 32;  // 2 waves along N

    // hoisted staging coordinates
    const int ar  = tid >> 3;             // A row (step 32 per iter)
    const int ac4 = (tid & 7) << 2;       // A col group of 4
    const int asc = aswz(ac4);            // swizzled LDS column
    const int bc  = tid & 63;             // B output-column (n)
    const int br4 = (tid >> 6) << 2;      // B k group of 4 (step 16 per iter)

    v8f acc[2][2];
#pragma unroll
    for (int i = 0; i < 2; ++i)
#pragma unroll
        for (int j = 0; j < 2; ++j)
#pragma unroll
            for (int r = 0; r < 8; ++r) acc[i][j][r] = 0.0f;

    // in-flight tile registers
    float4 ra[4];
    float  rbv[8];

    auto loadTiles = [&](int kk) {
#pragma unroll
        for (int it = 0; it < 4; ++it) {
            int r = ar + it * 32;
            ra[it] = *(const float4*)(A + (long long)(mBlock + r) * lda + (kk + ac4));
        }
        if (transB) {
#pragma unroll
            for (int it = 0; it < 2; ++it) {
                int r4 = br4 + it * 16;
                float4 v = *(const float4*)(Bm + (long long)(nBlock + bc) * ldb + (kk + r4));
                rbv[it * 4 + 0] = v.x; rbv[it * 4 + 1] = v.y;
                rbv[it * 4 + 2] = v.z; rbv[it * 4 + 3] = v.w;
            }
        } else {
#pragma unroll
            for (int it = 0; it < 2; ++it) {
                int r4 = br4 + it * 16;
                const float* p = Bm + (long long)(kk + r4) * ldb + (nBlock + bc);
                rbv[it * 4 + 0] = p[0];        rbv[it * 4 + 1] = p[ldb];
                rbv[it * 4 + 2] = p[2 * ldb];  rbv[it * 4 + 3] = p[3 * ldb];
            }
        }
    };

    auto storeTiles = [&](int buf) {
#pragma unroll
        for (int it = 0; it < 4; ++it) {
            int r = ar + it * 32;
            uint2 u; u.x = pk2(ra[it].x, ra[it].y); u.y = pk2(ra[it].z, ra[it].w);
            *(uint2*)&aT[buf][r][asc] = u;
        }
#pragma unroll
        for (int it = 0; it < 2; ++it) {
            int r4 = br4 + it * 16;
            uint2 u; u.x = pk2(rbv[it * 4 + 0], rbv[it * 4 + 1]);
                     u.y = pk2(rbv[it * 4 + 2], rbv[it * 4 + 3]);
            *(uint2*)&bT[buf][bc][aswz(r4)] = u;
        }
    };

    const int nk = K >> 5;                // K always multiple of 32
    loadTiles(0);                         // prologue: tile 0 -> regs

    for (int i = 0; i < nk; ++i) {
        const int buf = i & 1;
        storeTiles(buf);                  // regs -> lds[buf] (f32->f16 packed)
        __syncthreads();                  // ONE barrier per K-step

        if (i + 1 < nk) {
            loadTiles((i + 1) << 5);      // issue next tile; wait sinks past WMMAs
            if (i + 2 < nk) {
                int kp = (i + 2) << 5;    // prefetch tile after next
                __builtin_prefetch(A + (long long)(mBlock + (tid & 127)) * lda + kp, 0, 1);
                const float* pb = transB ? (Bm + (long long)(nBlock + bc) * ldb + kp)
                                         : (Bm + (long long)kp * ldb + (nBlock + bc));
                __builtin_prefetch(pb, 0, 1);
            }
        }

        // fragments: one contiguous 32B v16h load each
        const int kOff = (lane >> 4) << 4;
        v16h af[2], bf[2];
#pragma unroll
        for (int i2 = 0; i2 < 2; ++i2)
            af[i2] = *(const v16h*)(&aT[buf][wM + i2 * 16 + (lane & 15)][kOff]);
#pragma unroll
        for (int j2 = 0; j2 < 2; ++j2)
            bf[j2] = *(const v16h*)(&bT[buf][wN + j2 * 16 + (lane & 15)][kOff]);

#pragma unroll
        for (int i2 = 0; i2 < 2; ++i2)
#pragma unroll
            for (int j2 = 0; j2 < 2; ++j2)
                acc[i2][j2] = __builtin_amdgcn_wmma_f32_16x16x32_f16(
                    false, af[i2], false, bf[j2], (short)0, acc[i2][j2], false, false);
    }

    // ---- epilogue ----
#pragma unroll
    for (int i = 0; i < 2; ++i) {
#pragma unroll
        for (int j = 0; j < 2; ++j) {
            int mT = mBlock + wM + i * 16 + ((lane >> 4) << 3);
            int nT = nBlock + wN + j * 16 + (lane & 15);
            bool nOK = nT < N;
#pragma unroll
            for (int r = 0; r < 8; ++r) {
                int m = mT + r;
                float v = acc[i][j][r] * alpha;
                if (bias)    v += bias[nT];                           // N mult of 64 when used
                if (biasMat) v += biasMat[(long long)m * ldbm + nT];  // padded buffer
                if (act == 1)      v = 0.5f * v * (1.0f + erff(v * 0.70710678118f));
                else if (act == 2) v = fmaxf(v, 0.0f);
                v = nOK ? v : 0.0f;       // zero pad columns (scores K-padding)
                if (m < M) C[(long long)m * ldc + nT] = v;            // predicated store
            }
        }
    }
}

// ---------------------------------------------------------------------------
// x = x0 + pos_embed (broadcast over batch)
// ---------------------------------------------------------------------------
__global__ void add_pos_kernel(const float* __restrict__ x0,
                               const float* __restrict__ pos,
                               float* __restrict__ x, long long total)
{
    long long i = (long long)blockIdx.x * blockDim.x + threadIdx.x;
    if (i >= total) return;
    x[i] = x0[i] + pos[i % (long long)(TT * DD)];
}

// ---------------------------------------------------------------------------
// Relative-position bias: bmat[q][k] = rel_pos[k-q+T-1] . rpe_w  (ld = TPAD)
// ---------------------------------------------------------------------------
__global__ void rpe_bias_kernel(const float* __restrict__ rel_pos,
                                const float* __restrict__ rpe_w,
                                float* __restrict__ bmat)
{
    int k = blockIdx.x * blockDim.x + threadIdx.x;
    int q = blockIdx.y;
    if (k >= TT) return;
    const float* rp = rel_pos + (long long)(k - q + (TT - 1)) * HDD;
    float s = 0.0f;
#pragma unroll
    for (int d = 0; d < HDD; ++d) s += rp[d] * rpe_w[d];
    bmat[q * TPAD + k] = s;
}

// ---------------------------------------------------------------------------
// Row softmax over TT cols (one 256-thread block per row, in place)
// scores layout: (b*h) slices of SROWS x TPAD
// ---------------------------------------------------------------------------
__global__ __launch_bounds__(256)
void softmax_kernel(float* __restrict__ p)
{
    __shared__ float red[256];
    int id = blockIdx.x;                 // 0 .. BB*HH*TT-1
    int bh = id / TT, r = id - bh * TT;
    float* row = p + (long long)bh * (SROWS * TPAD) + (long long)r * TPAD;
    int tid = threadIdx.x;

    float m = -1e30f;
    for (int j = tid; j < TT; j += 256) m = fmaxf(m, row[j]);
    red[tid] = m; __syncthreads();
    for (int s = 128; s > 0; s >>= 1) {
        if (tid < s) red[tid] = fmaxf(red[tid], red[tid + s]);
        __syncthreads();
    }
    m = red[0]; __syncthreads();

    float sum = 0.0f;
    for (int j = tid; j < TT; j += 256) {
        float e = __expf(row[j] - m);
        row[j] = e;
        sum += e;
    }
    red[tid] = sum; __syncthreads();
    for (int s = 128; s > 0; s >>= 1) {
        if (tid < s) red[tid] += red[tid + s];
        __syncthreads();
    }
    float inv = 1.0f / red[0]; __syncthreads();
    for (int j = tid; j < TT; j += 256) row[j] *= inv;
}

// ---------------------------------------------------------------------------
// x = LayerNorm(x + h) * g + b     (one block per row, D = 512)
// ---------------------------------------------------------------------------
__global__ __launch_bounds__(256)
void add_ln_kernel(float* __restrict__ x, const float* __restrict__ h,
                   const float* __restrict__ g, const float* __restrict__ b)
{
    __shared__ float red[256];
    long long base = (long long)blockIdx.x * DD;
    int tid = threadIdx.x;
    float v0 = x[base + tid]       + h[base + tid];
    float v1 = x[base + tid + 256] + h[base + tid + 256];

    red[tid] = v0 + v1; __syncthreads();
    for (int s = 128; s > 0; s >>= 1) {
        if (tid < s) red[tid] += red[tid + s];
        __syncthreads();
    }
    float mean = red[0] * (1.0f / DD); __syncthreads();

    float d0 = v0 - mean, d1 = v1 - mean;
    red[tid] = d0 * d0 + d1 * d1; __syncthreads();
    for (int s = 128; s > 0; s >>= 1) {
        if (tid < s) red[tid] += red[tid + s];
        __syncthreads();
    }
    float rs = rsqrtf(red[0] * (1.0f / DD) + LN_EPS);

    x[base + tid]       = d0 * rs * g[tid]       + b[tid];
    x[base + tid + 256] = d1 * rs * g[tid + 256] + b[tid + 256];
}

// ---------------------------------------------------------------------------
// shift[r] = h3[r,:] . mlp_w2 + mlp_b2     (h3 already ReLU'd by GEMM epilogue)
// ---------------------------------------------------------------------------
__global__ void shift_kernel(const float* __restrict__ h3,
                             const float* __restrict__ w2,
                             const float* __restrict__ b2,
                             float* __restrict__ shift)
{
    int r = blockIdx.x * blockDim.x + threadIdx.x;
    if (r >= ROWS) return;
    float s = b2[0];
    const float* hr = h3 + (long long)r * 64;
#pragma unroll
    for (int j = 0; j < 64; ++j) s += hr[j] * w2[j];
    shift[r] = s;
}

// ---------------------------------------------------------------------------
// Bilinear warp of x0 by predicted flow
// ---------------------------------------------------------------------------
__global__ void warp_kernel(const float* __restrict__ x0,
                            const float* __restrict__ shift,
                            const int* __restrict__ max_shift,
                            float* __restrict__ warped)
{
    long long gid = (long long)blockIdx.x * blockDim.x + threadIdx.x;
    const long long total = (long long)BB * TT * DD;
    if (gid >= total) return;
    int l = (int)(gid % DD);
    long long bt = gid / DD;
    int t = (int)(bt % TT);
    int b = (int)(bt / TT);

    float ms   = (float)max_shift[0];
    float flow = shift[b * TT + t] * ms * (1.0f / (DD / 2));
    float gx = 2.0f * ((float)l / (DD - 1) - 0.5f) + flow;
    float gy = 2.0f * ((float)t / (TT - 1) - 0.5f);
    float ix = fminf(fmaxf(((gx + 1.0f) * DD - 1.0f) * 0.5f, 0.0f), (float)(DD - 1));
    float iy = fminf(fmaxf(((gy + 1.0f) * TT - 1.0f) * 0.5f, 0.0f), (float)(TT - 1));

    float ix0f = floorf(ix), wx = ix - ix0f;
    float iy0f = floorf(iy), wy = iy - iy0f;
    int ix0 = min(max((int)ix0f, 0), DD - 1);
    int ix1 = min(ix0 + 1, DD - 1);
    int iy0 = min(max((int)iy0f, 0), TT - 1);
    int iy1 = min(iy0 + 1, TT - 1);

    const float* xb = x0 + (long long)b * TT * DD;
    float v00 = xb[(long long)iy0 * DD + ix0];
    float v01 = xb[(long long)iy0 * DD + ix1];
    float v10 = xb[(long long)iy1 * DD + ix0];
    float v11 = xb[(long long)iy1 * DD + ix1];
    warped[gid] = (1.0f - wy) * ((1.0f - wx) * v00 + wx * v01) +
                  wy          * ((1.0f - wx) * v10 + wx * v11);
}

// ---------------------------------------------------------------------------
// Host-side GEMM launch helper
// ---------------------------------------------------------------------------
static void launch_gemm(hipStream_t stream,
                        const float* A, const float* Bm,
                        const float* bias, const float* biasMat, float* C,
                        int M, int N, int K,
                        int lda, int ldb, int ldc, int ldbm,
                        long long sAb, long long sAh,
                        long long sBb, long long sBh,
                        long long sCb, long long sCh,
                        int batches, int nbatch2, int transB, int act, float alpha)
{
    dim3 grid((N + GTN - 1) / GTN, (M + GTM - 1) / GTM, batches);
    gemm_wmma_kernel<<<grid, 256, 0, stream>>>(A, Bm, bias, biasMat, C,
                                               M, N, K, lda, ldb, ldc, ldbm,
                                               sAb, sAh, sBb, sBh, sCb, sCh,
                                               nbatch2, transB, act, alpha);
}

// ---------------------------------------------------------------------------
// kernel_launch
// ---------------------------------------------------------------------------
extern "C" void kernel_launch(void* const* d_in, const int* in_sizes, int n_in,
                              void* d_out, int out_size, void* d_ws, size_t ws_size,
                              hipStream_t stream)
{
    const float* x0        = (const float*)d_in[0];
    const int*   max_shift = (const int*)  d_in[1];
    const float* pos_embed = (const float*)d_in[2];
    const float* qkv_w     = (const float*)d_in[3];
    const float* qkv_b     = (const float*)d_in[4];
    const float* out_w     = (const float*)d_in[5];
    const float* out_b     = (const float*)d_in[6];
    const float* rel_pos   = (const float*)d_in[7];
    const float* rpe_w     = (const float*)d_in[8];
    const float* ln1_g     = (const float*)d_in[9];
    const float* ln1_b     = (const float*)d_in[10];
    const float* ffn_w1    = (const float*)d_in[11];
    const float* ffn_b1    = (const float*)d_in[12];
    const float* ffn_w2    = (const float*)d_in[13];
    const float* ffn_b2    = (const float*)d_in[14];
    const float* ln2_g     = (const float*)d_in[15];
    const float* ln2_b     = (const float*)d_in[16];
    const float* mlp_w1    = (const float*)d_in[17];
    const float* mlp_b1    = (const float*)d_in[18];
    const float* mlp_w2    = (const float*)d_in[19];
    const float* mlp_b2    = (const float*)d_in[20];

    // workspace layout (floats) — all buffers padded for guard-free staging
    float* ws = (float*)d_ws;
    size_t off = 0;
    float* xbuf   = ws + off; off += (size_t)ROWS_P * DD;               // padded rows
    float* qkvb   = ws + off; off += (size_t)ROWS_P * 3 * DD;
    float* scores = ws + off; off += (size_t)BB * HH * SROWS * TPAD;    // padded slices
    float* attnb  = ws + off; off += (size_t)ROWS_P * DD;
    float* h1     = ws + off; off += (size_t)ROWS_P * DD;
    float* h2     = ws + off; off += (size_t)ROWS_P * FFN;
    float* bmat   = ws + off; off += (size_t)SROWS * TPAD;              // padded bias mat
    float* h3     = ws + off; off += (size_t)ROWS_P * 64;

    float* shiftbuf = (float*)d_out;           // first  (B*T) floats
    float* warped   = (float*)d_out + ROWS;    // then   (B*T*D) floats

    const long long total = (long long)ROWS * DD;
    const float scale = 0.125f;                // hd^-0.5 = 1/sqrt(64)

    // x = x0 + pos_embed
    add_pos_kernel<<<(int)((total + 255) / 256), 256, 0, stream>>>(x0, pos_embed, xbuf, total);

    for (int i = 0; i < DEPTH; ++i) {
        const float* wq  = qkv_w  + (size_t)i * DD * 3 * DD;
        const float* bq  = qkv_b  + (size_t)i * 3 * DD;
        const float* wo  = out_w  + (size_t)i * DD * DD;
        const float* bo  = out_b  + (size_t)i * DD;
        const float* rpi = rel_pos + (size_t)i * (2 * TT - 1) * HDD;
        const float* rwi = rpe_w  + (size_t)i * HDD;
        const float* w1  = ffn_w1 + (size_t)i * DD * FFN;
        const float* b1  = ffn_b1 + (size_t)i * FFN;
        const float* w2  = ffn_w2 + (size_t)i * FFN * DD;
        const float* b2  = ffn_b2 + (size_t)i * DD;

        // qkv = x @ Wqkv + b       (5792 x 1536, K=512)
        launch_gemm(stream, xbuf, wq, bq, nullptr, qkvb,
                    ROWS, 3 * DD, DD, DD, 3 * DD, 3 * DD, 0,
                    0, 0, 0, 0, 0, 0, 1, 1, 0, 0, 1.0f);

        // relative position bias matrix (181 x 181 in 256x192 padded buffer)
        rpe_bias_kernel<<<dim3(1, TT), 256, 0, stream>>>(rpi, rwi, bmat);

        // scores[b,h] = scale * Q Kᵀ + bmat   (batched 181x181, K=64)
        // C slices padded SROWS x TPAD; pad columns written as ZERO.
        launch_gemm(stream,
                    qkvb,            /* Q part, +h*64 via sAh */
                    qkvb + DD,       /* K part */
                    nullptr, bmat, scores,
                    TT, TT, HDD,
                    3 * DD, 3 * DD, TPAD, TPAD,
                    (long long)TT * 3 * DD, 64,
                    (long long)TT * 3 * DD, 64,
                    (long long)HH * SROWS * TPAD, (long long)SROWS * TPAD,
                    BB * HH, HH, /*transB=*/1, 0, scale);

        // softmax rows (pad columns stay zero)
        softmax_kernel<<<BB * HH * TT, 256, 0, stream>>>(scores);

        // attn[b,:,h*64:] = P @ V   (batched 181x64, K=192: zero A-cols kill V garbage)
        launch_gemm(stream,
                    scores, qkvb + 2 * DD, nullptr, nullptr, attnb,
                    TT, HDD, TPAD,
                    TPAD, 3 * DD, DD, 0,
                    (long long)HH * SROWS * TPAD, (long long)SROWS * TPAD,
                    (long long)TT * 3 * DD, 64,
                    (long long)TT * DD, 64,
                    BB * HH, HH, 0, 0, 1.0f);

        // out projection
        launch_gemm(stream, attnb, wo, bo, nullptr, h1,
                    ROWS, DD, DD, DD, DD, DD, 0,
                    0, 0, 0, 0, 0, 0, 1, 1, 0, 0, 1.0f);

        // x = LN(x + attn_out)
        add_ln_kernel<<<ROWS, 256, 0, stream>>>(xbuf, h1,
                                                ln1_g + (size_t)i * DD,
                                                ln1_b + (size_t)i * DD);

        // FFN1 + exact GELU
        launch_gemm(stream, xbuf, w1, b1, nullptr, h2,
                    ROWS, FFN, DD, DD, FFN, FFN, 0,
                    0, 0, 0, 0, 0, 0, 1, 1, 0, 1, 1.0f);

        // FFN2
        launch_gemm(stream, h2, w2, b2, nullptr, h1,
                    ROWS, DD, FFN, FFN, DD, DD, 0,
                    0, 0, 0, 0, 0, 0, 1, 1, 0, 0, 1.0f);

        // x = LN(x + ffn_out)
        add_ln_kernel<<<ROWS, 256, 0, stream>>>(xbuf, h1,
                                                ln2_g + (size_t)i * DD,
                                                ln2_b + (size_t)i * DD);
    }

    // head MLP: h3 = relu(x @ mlp_w1 + b1)    (5792 x 64)
    launch_gemm(stream, xbuf, mlp_w1, mlp_b1, nullptr, h3,
                ROWS, 64, DD, DD, 64, 64, 0,
                0, 0, 0, 0, 0, 0, 1, 1, 0, 2, 1.0f);

    // shift = h3 @ mlp_w2 + b2
    shift_kernel<<<(ROWS + 255) / 256, 256, 0, stream>>>(h3, mlp_w2, mlp_b2, shiftbuf);

    // warped output
    warp_kernel<<<(int)((total + 255) / 256), 256, 0, stream>>>(x0, shiftbuf, max_shift, warped);
}
